// HeteroGNN_SAGE_79448305041987
// MI455X (gfx1250) — matrix operand order repaired and checked by reading, not compiled, not run on hardware
//
#include <hip/hip_runtime.h>
#include <hip/hip_bf16.h>
#include <stdint.h>

// ---------------------------------------------------------------------------
// HeteroGNN SAGE on MI455X (gfx1250): bf16 WMMA GEMMs + atomic edge scatter.
// Round 2: LDS tiles stored pre-converted bf16 in fragment-contiguous layout
// so the WMMA K-loop is pure ds_load_b128 + v_wmma (no per-element cvt VALU).
// ---------------------------------------------------------------------------

typedef __bf16 bf16_t;
typedef __attribute__((ext_vector_type(16))) __bf16 v16bf;
typedef __attribute__((ext_vector_type(8)))  __bf16 v8bf;
typedef __attribute__((ext_vector_type(8)))  float  v8f;

#define HD    128   // hidden / feature dim
#define SWPC  136   // padded LDS stride for W, column-major (bf16 elems, %8==0)
#define SAPB  136   // padded LDS stride for A (bf16 elems, %8==0)
#define ROWS   64   // output rows per block
#define PCHUNK 256  // rows per pooling block

static __device__ __forceinline__ bf16_t f2bf(float x) {
  // round-to-nearest-even float -> bf16
  unsigned u = __builtin_bit_cast(unsigned, x);
  u += 0x7FFFu + ((u >> 16) & 1u);
  unsigned short h = (unsigned short)(u >> 16);
  return __builtin_bit_cast(bf16_t, h);
}

// ---------------------------------------------------------------------------
__global__ void hg_zero_kernel(float* __restrict__ p, long n) {
  long i = (long)blockIdx.x * blockDim.x + threadIdx.x;
  long stride = (long)gridDim.x * blockDim.x;
  for (; i < n; i += stride) p[i] = 0.0f;
}

// One wave per edge: 32 lanes x float4 = 128 features. Atomic scatter-add of
// the source row into agg[dst], plus one degree count per edge.
__global__ void hg_scatter_kernel(const float* __restrict__ feat,
                                  const int* __restrict__ src,
                                  const int* __restrict__ dst,
                                  float* __restrict__ agg,
                                  float* __restrict__ cnt, int E) {
  long tid = (long)blockIdx.x * blockDim.x + threadIdx.x;
  int e = (int)(tid >> 5);
  int q = (int)(tid & 31);
  if (e >= E) return;
  int s = src[e];
  int d = dst[e];
  const float4 v = *(const float4*)(feat + (size_t)s * HD + q * 4);
  float* base = agg + (size_t)d * HD + q * 4;
  atomicAdd(base + 0, v.x);
  atomicAdd(base + 1, v.y);
  atomicAdd(base + 2, v.z);
  atomicAdd(base + 3, v.w);
  if (q == 0) atomicAdd(cnt + d, 1.0f);
}

// ---------------------------------------------------------------------------
// C[n,128] (first ? = : +=) (A (/cnt)) @ (W1 (+W2)) (+ b1 + b2) (leaky if act)
// Block: 256 threads = 8 waves. Block tile: 64 rows x 128 cols.
// Wave w owns column tile w*16..w*16+15 and 4 row-tiles of 16.
// K loop: 4 steps of v_wmma_f32_16x16x32_bf16 (f32 accumulate) x 4 row tiles.
__global__ __launch_bounds__(256) void hg_gemm_kernel(
    const float* __restrict__ A, const float* __restrict__ cnt,
    const float* __restrict__ W1, const float* __restrict__ W2,
    const float* __restrict__ b1, const float* __restrict__ b2,
    float* __restrict__ C, int n, int first, int act) {
  // W column-major: sW[col * SWPC + k]  (bf16, ~34.8 KB)
  __shared__ __align__(16) bf16_t sW[HD * SWPC];
  // A row-major: sA[row * SAPB + k]     (bf16, ~17.4 KB)
  __shared__ __align__(16) bf16_t sA[ROWS * SAPB];

  const int tid = threadIdx.x;

  // Stage W (+W2) into LDS as bf16, transposed to column-major.
  for (int i = tid; i < HD * HD; i += 256) {
    int r = i >> 7, c = i & 127;     // W[r][c], global read coalesced
    float w = W1[i];
    if (W2) w += W2[i];
    sW[c * SWPC + r] = f2bf(w);
  }
  // Stage 64 A rows into LDS as bf16 with optional mean division.
  const int row0 = blockIdx.x * ROWS;
  for (int i = tid; i < ROWS * HD; i += 256) {
    int r = i >> 7, k = i & 127;
    int row = row0 + r;
    float v = 0.0f;
    if (row < n) {
      v = A[(size_t)row * HD + k];
      if (cnt) v /= fmaxf(cnt[row], 1.0f);
    }
    sA[r * SAPB + k] = f2bf(v);
  }
  __syncthreads();

  const int wave = tid >> 5;
  const int lane = tid & 31;
  const int half = lane >> 4;   // lane half selects K sub-range
  const int l16  = lane & 15;
  const int col  = wave * 16 + l16;

  v8f acc[4] = {v8f{}, v8f{}, v8f{}, v8f{}};

#pragma unroll
  for (int kk = 0; kk < HD; kk += 32) {
    // B fragment (32x16): bfrag[j] = W[kk + half*16 + j][col], j = 0..15
    // -> 16 consecutive bf16 in column-major sW: two 16B LDS loads.
    const bf16_t* bp = &sW[col * SWPC + kk + half * 16];
    v8bf b0 = *(const v8bf*)(bp);
    v8bf b1v = *(const v8bf*)(bp + 8);
    v16bf bfrag = __builtin_shufflevector(
        b0, b1v, 0, 1, 2, 3, 4, 5, 6, 7, 8, 9, 10, 11, 12, 13, 14, 15);
#pragma unroll
    for (int t = 0; t < 4; ++t) {
      // A fragment (16x32): elems 0..15 = K kk+half*16+0..15.
      const bf16_t* ap = &sA[(t * 16 + l16) * SAPB + kk + half * 16];
      v8bf a0 = *(const v8bf*)(ap);
      v8bf a1 = *(const v8bf*)(ap + 8);
      v16bf afrag = __builtin_shufflevector(
          a0, a1, 0, 1, 2, 3, 4, 5, 6, 7, 8, 9, 10, 11, 12, 13, 14, 15);
      acc[t] = __builtin_amdgcn_wmma_f32_16x16x32_bf16(
          false, afrag, false, bfrag, (short)0, acc[t], false, false);
    }
  }

  // Epilogue. C/D layout: VGPR r -> M = r + 8*half, N = l16.
  float bias = 0.0f;
  if (first) {
    if (b1) bias += b1[col];
    if (b2) bias += b2[col];
  }
#pragma unroll
  for (int t = 0; t < 4; ++t) {
#pragma unroll
    for (int r = 0; r < 8; ++r) {
      int row = row0 + t * 16 + half * 8 + r;
      if (row < n) {
        size_t idx = (size_t)row * HD + col;
        float v = acc[t][r];
        v += first ? bias : C[idx];
        if (act) v = (v >= 0.0f) ? v : 0.01f * v;
        C[idx] = v;
      }
    }
  }
}

// ---------------------------------------------------------------------------
// Segment-sum pooling over sorted batch ids: per-chunk register accumulation,
// atomic flush only at graph boundaries (~2 per chunk).
__global__ void hg_pool_kernel(const float* __restrict__ h,
                               const int* __restrict__ batch,
                               float* __restrict__ pooled,  // [G, 256]
                               int n, int coloff) {
  int f = threadIdx.x;  // 0..127
  int r0 = blockIdx.x * PCHUNK;
  if (r0 >= n) return;
  int r1 = r0 + PCHUNK;
  if (r1 > n) r1 = n;
  float acc = 0.0f;
  int cur = batch[r0];
  for (int r = r0; r < r1; ++r) {
    int b = batch[r];
    if (b != cur) {
      atomicAdd(&pooled[(size_t)cur * 256 + coloff + f], acc);
      acc = 0.0f;
      cur = b;
    }
    acc += h[(size_t)r * HD + f];
  }
  atomicAdd(&pooled[(size_t)cur * 256 + coloff + f], acc);
}

// out[g] = (pooled[g,:] @ mlp_W + mlp_b) @ lin_W + lin_b   (64x256x128, tiny)
__global__ void hg_head_kernel(const float* __restrict__ pooled,
                               const float* __restrict__ mlpW,
                               const float* __restrict__ mlpb,
                               const float* __restrict__ linW,
                               const float* __restrict__ linb,
                               float* __restrict__ out) {
  __shared__ float red[128];
  int g = blockIdx.x, j = threadIdx.x;
  const float* gr = pooled + (size_t)g * 256;
  float t = mlpb[j];
  for (int k = 0; k < 256; ++k) t += gr[k] * mlpW[k * HD + j];
  t *= linW[j];
  red[j] = t;
  __syncthreads();
  for (int s = 64; s > 0; s >>= 1) {
    if (j < s) red[j] += red[j + s];
    __syncthreads();
  }
  if (j == 0) out[g] = red[0] + linb[0];
}

// ---------------------------------------------------------------------------
extern "C" void kernel_launch(void* const* d_in, const int* in_sizes, int n_in,
                              void* d_out, int out_size, void* d_ws, size_t ws_size,
                              hipStream_t stream) {
  (void)in_sizes; (void)n_in; (void)out_size; (void)ws_size;

  const float* x_a    = (const float*)d_in[0];
  const float* x_b    = (const float*)d_in[1];
  const int*   aa_src = (const int*)d_in[2];
  const int*   aa_dst = (const int*)d_in[3];
  const int*   ab_src = (const int*)d_in[4];
  const int*   ab_dst = (const int*)d_in[5];
  const int*   ba_src = (const int*)d_in[6];
  const int*   ba_dst = (const int*)d_in[7];
  const int*   batch_a = (const int*)d_in[8];
  const int*   batch_b = (const int*)d_in[9];
  const float* Wl    = (const float*)d_in[10];  // [2][3][128][128]
  const float* bl    = (const float*)d_in[11];  // [2][3][128]
  const float* Wr    = (const float*)d_in[12];  // [2][3][128][128]
  const float* mlp_W = (const float*)d_in[13];  // [256][128]
  const float* mlp_b = (const float*)d_in[14];  // [128]
  const float* lin_W = (const float*)d_in[15];  // [128][1]
  const float* lin_b = (const float*)d_in[16];  // [1]

  const int NA = 100000, NB = 100000, E = 640000, G = 64;

  // Workspace layout (~257 MB)
  float* p = (float*)d_ws;
  float* ha1 = p; p += (size_t)NA * HD;
  float* hb1 = p; p += (size_t)NB * HD;
  float* ha2 = p; p += (size_t)NA * HD;
  float* hb2 = p; p += (size_t)NB * HD;
  float* agg = p; p += (size_t)NA * HD;   // NA == NB, shared serially
  float* cnt = p; p += NA;
  float* pooled = p; p += (size_t)G * 256;

  const dim3 b256(256);
  const int zgrid = 2048;
  const int sgrid = (int)(((long)E * 32 + 255) / 256);
  const int ggridA = (NA + ROWS - 1) / ROWS;
  const int ggridB = (NB + ROWS - 1) / ROWS;

  auto Woff = [](int l, int r) { return (size_t)(l * 3 + r) * HD * HD; };
  auto boff = [](int l, int r) { return (size_t)(l * 3 + r) * HD; };

  struct Lay { const float* a; const float* b; float* oa; float* ob; };
  const Lay lay[2] = {{x_a, x_b, ha1, hb1}, {ha1, hb1, ha2, hb2}};

  for (int l = 0; l < 2; ++l) {
    const float* pa = lay[l].a;
    const float* pb = lay[l].b;
    float* oa = lay[l].oa;
    float* ob = lay[l].ob;

    // ---- dst = a : relation aa (a->a) ----
    hg_zero_kernel<<<zgrid, b256, 0, stream>>>(agg, (long)NA * HD);
    hg_zero_kernel<<<256, b256, 0, stream>>>(cnt, NA);
    hg_scatter_kernel<<<sgrid, b256, 0, stream>>>(pa, aa_src, aa_dst, agg, cnt, E);
    hg_gemm_kernel<<<ggridA, b256, 0, stream>>>(
        agg, cnt, Wl + Woff(l, 0), nullptr, bl + boff(l, 0), bl + boff(l, 2),
        oa, NA, /*first=*/1, /*act=*/0);

    // ---- dst = a : relation ba (b->a) ----
    hg_zero_kernel<<<zgrid, b256, 0, stream>>>(agg, (long)NA * HD);
    hg_zero_kernel<<<256, b256, 0, stream>>>(cnt, NA);
    hg_scatter_kernel<<<sgrid, b256, 0, stream>>>(pb, ba_src, ba_dst, agg, cnt, E);
    hg_gemm_kernel<<<ggridA, b256, 0, stream>>>(
        agg, cnt, Wl + Woff(l, 2), nullptr, nullptr, nullptr,
        oa, NA, /*first=*/0, /*act=*/0);

    // ---- dst = a : self term, Wr0 + Wr2 fused; leaky epilogue ----
    hg_gemm_kernel<<<ggridA, b256, 0, stream>>>(
        pa, nullptr, Wr + Woff(l, 0), Wr + Woff(l, 2), nullptr, nullptr,
        oa, NA, /*first=*/0, /*act=*/1);

    // ---- dst = b : relation ab (a->b) ----
    hg_zero_kernel<<<zgrid, b256, 0, stream>>>(agg, (long)NB * HD);
    hg_zero_kernel<<<256, b256, 0, stream>>>(cnt, NB);
    hg_scatter_kernel<<<sgrid, b256, 0, stream>>>(pa, ab_src, ab_dst, agg, cnt, E);
    hg_gemm_kernel<<<ggridB, b256, 0, stream>>>(
        agg, cnt, Wl + Woff(l, 1), nullptr, bl + boff(l, 1), nullptr,
        ob, NB, /*first=*/1, /*act=*/0);

    // ---- dst = b : self term; leaky epilogue ----
    hg_gemm_kernel<<<ggridB, b256, 0, stream>>>(
        pb, nullptr, Wr + Woff(l, 1), nullptr, nullptr, nullptr,
        ob, NB, /*first=*/0, /*act=*/1);
  }

  // ---- pooling + head ----
  hg_zero_kernel<<<16, b256, 0, stream>>>(pooled, (long)G * 256);
  hg_pool_kernel<<<(NA + PCHUNK - 1) / PCHUNK, 128, 0, stream>>>(
      ha2, batch_a, pooled, NA, /*coloff=*/0);
  hg_pool_kernel<<<(NB + PCHUNK - 1) / PCHUNK, 128, 0, stream>>>(
      hb2, batch_b, pooled, NB, /*coloff=*/HD);
  hg_head_kernel<<<G, 128, 0, stream>>>(pooled, mlp_W, mlp_b, lin_W, lin_b,
                                        (float*)d_out);
}
